// MemoryBank_45019847196883
// MI455X (gfx1250) — matrix operand order repaired and checked by reading, not compiled, not run on hardware
//
#include <hip/hip_runtime.h>
#include <hip/hip_bf16.h>

typedef __attribute__((ext_vector_type(16))) _Float16 v16h;
typedef __attribute__((ext_vector_type(8)))  _Float16 v8h;
typedef __attribute__((ext_vector_type(4)))  _Float16 v4h;
typedef __attribute__((ext_vector_type(8)))  float    v8f;
typedef __attribute__((ext_vector_type(4)))  float    v4f;
typedef __attribute__((ext_vector_type(4)))  unsigned int v4u;
typedef __attribute__((ext_vector_type(8)))  int      v8i;
typedef __attribute__((ext_vector_type(4)))  int      v4i;

#define DIMD     128
#define WAVES    8
#define BLOCK    256
#define QROWS_WG 128     // query rows per workgroup (16 per wave)
#define KT2      32      // queue rows staged per TDM round (2 sub-tiles of 16)
#define LDP      136     // padded LDS row stride in halves (272B; TDM pad does this)
#define NSPLIT   64      // K-dimension splits
#define TOPK     16

// ---------------------------------------------------------------------------
// intra-wave LDS producer->consumer sync (slow path is wave-divergent).
__device__ __forceinline__ void wave_lds_sync() {
    __builtin_amdgcn_wave_barrier();
    asm volatile("s_wait_dscnt 0" ::: "memory");
    __builtin_amdgcn_wave_barrier();
}

// ---------------------------------------------------------------------------
// TDM: DMA one (KT2 x 128) f16 tile from global into LDS with row padding.
// D# per CDNA5 ISA 8.3-8.6:
//   group0: count=1 | lds_addr | global_addr(57b) | type=2
//   group1: data_size=2B, pad_enable, pad_interval=64dw(code 5),
//           pad_amount=4dw(code 3) -> LDS row stride 256+16B = 272B = LDP halves
//           tensor_dim0=128, tensor_dim1=KT2, tile_dim0=128, tile_dim1=KT2,
//           tensor_dim0_stride=128
// Toolchain here is the clang-23 6-arg builtin:
//   (uint32x4 g0, int32x8 g1, int32x4 g2, int32x4 g3, int32x8 pad, i32 cpol)
__device__ __forceinline__ void tdm_load_tile(const _Float16* gsrc,
                                              unsigned int lds_byte) {
    const unsigned long long ga = (unsigned long long)(uintptr_t)gsrc;
    v4u g0;
    g0[0] = 1u;                                            // count=1
    g0[1] = lds_byte;                                      // lds_addr
    g0[2] = (unsigned int)ga;                              // global_addr lo
    g0[3] = ((unsigned int)(ga >> 32) & 0x01FFFFFFu) | (2u << 30); // hi | type=2
    v8i g1;
    g1[0] = (int)((1u << 16) | (1u << 20) | (5u << 22) | (3u << 25));
    g1[1] = (int)(128u << 16);          // tensor_dim0[15:0] in bits 63:48
    g1[2] = (int)((unsigned)KT2 << 16); // tensor_dim1[15:0] in bits 95:80
    g1[3] = (int)(128u << 16);          // tile_dim0 in bits 127:112
    g1[4] = KT2;                        // tile_dim1 (tile_dim2=0)
    g1[5] = 128;                        // tensor_dim0_stride lo32
    g1[6] = 0;
    g1[7] = 0;
    const v4i z4 = {0, 0, 0, 0};
    const v8i z8 = {0, 0, 0, 0, 0, 0, 0, 0};
    __builtin_amdgcn_tensor_load_to_lds(g0, g1, z4, z4, z8, 0);
}

// ---------------------------------------------------------------------------
// Kernel 1: L2-normalize query rows, convert to f16. One wave per row.
__global__ __launch_bounds__(BLOCK) void mb_norm(const float* __restrict__ q,
                                                 _Float16* __restrict__ qf16,
                                                 int n) {
    const int row  = blockIdx.x * (BLOCK / 32) + (threadIdx.x >> 5);
    const int lane = threadIdx.x & 31;
    if (row >= n) return;
    v4f x = *(const v4f*)(q + (size_t)row * DIMD + lane * 4);
    float ss = x[0]*x[0] + x[1]*x[1] + x[2]*x[2] + x[3]*x[3];
#pragma unroll
    for (int m = 16; m >= 1; m >>= 1) ss += __shfl_xor(ss, m, 32);
    const float s = rsqrtf(fmaxf(ss, 1e-12f));
    v4h h;
#pragma unroll
    for (int i = 0; i < 4; ++i) h[i] = (_Float16)(x[i] * s);
    *(v4h*)(qf16 + (size_t)row * DIMD + lane * 4) = h;
}

// ---------------------------------------------------------------------------
// Kernel 1b: bulk queue f32 -> f16 (once; queue then lives f16 in L2).
__global__ __launch_bounds__(BLOCK) void mb_cvt(const float* __restrict__ src,
                                                _Float16* __restrict__ dst,
                                                size_t nelem) {
    const size_t i = ((size_t)blockIdx.x * BLOCK + threadIdx.x) * 8;
    if (i >= nelem) return;
    v4f a = *(const v4f*)(src + i);
    v4f b = *(const v4f*)(src + i + 4);
    v8h h;
#pragma unroll
    for (int j = 0; j < 4; ++j) { h[j] = (_Float16)a[j]; h[4 + j] = (_Float16)b[j]; }
    *(v8h*)(dst + i) = h;
}

// ---------------------------------------------------------------------------
// Kernel 2: fused f16-WMMA GEMM + streaming top-16 over a K split.
// grid: (NSPLIT, n/128), block: 256 (8 waves x 16 query rows).
// Queue tiles staged by TDM (double-buffered, wave 0 drives the DMA).
__global__ __launch_bounds__(BLOCK) void mb_main(
    const _Float16* __restrict__ qf16,     // [n][128] normalized f16
    const _Float16* __restrict__ queue16,  // [K][128] f16
    float* __restrict__ pval,              // [n][NSPLIT*16]
    int*   __restrict__ pidx,
    int n, int K) {
    __shared__ __align__(16) _Float16 tile[2][KT2 * LDP];
    __shared__ float simtile[WAVES][16][17];
    __shared__ float topv[WAVES][16][TOPK];
    __shared__ int   topi[WAVES][16][TOPK];
    __shared__ float thr [WAVES][16];
    __shared__ int   mpos[WAVES][16];

    const int tid  = threadIdx.x;
    const int w    = tid >> 5;
    const int lane = tid & 31;
    const int l16  = lane & 15;
    const int half = lane >> 4;

    const int kchunk = K / NSPLIT;           // queue rows per split
    const int rounds = kchunk / KT2;         // TDM rounds per split
    const int qbase  = blockIdx.y * QROWS_WG + w * 16;
    const int kbase  = blockIdx.x * kchunk;

    if (lane < 16) {
        thr[w][lane]  = -1e30f;
        mpos[w][lane] = 0;
#pragma unroll
        for (int j = 0; j < TOPK; ++j) { topv[w][lane][j] = -1e30f; topi[w][lane][j] = 0; }
    }

    // A fragments: ISA 16-bit A 16x32 layout. lane: M=l16, K-base = half*8.
    v16h afrag[4];
    {
        const _Float16* qp = qf16 + (size_t)(qbase + l16) * DIMD + half * 8;
#pragma unroll
        for (int c = 0; c < 4; ++c) {
            v8h lo = *(const v8h*)(qp + 32 * c);
            v8h hi = *(const v8h*)(qp + 32 * c + 16);
            afrag[c] = __builtin_shufflevector(lo, hi,
                0,1,2,3,4,5,6,7,8,9,10,11,12,13,14,15);
        }
    }

    // per-lane thresholds matching C/D layout: row(v) = v + half*8
    float T[8];
#pragma unroll
    for (int v = 0; v < 8; ++v) T[v] = -1e30f;

    const unsigned int lds0 = (unsigned int)(uintptr_t)(&tile[0][0]);
    const unsigned int lds1 = (unsigned int)(uintptr_t)(&tile[1][0]);

    // prologue: DMA round 0 into buffer 0
    if (w == 0) tdm_load_tile(queue16 + (size_t)kbase * DIMD, lds0);

    for (int r = 0; r < rounds; ++r) {
        if (w == 0) {
            if (r + 1 < rounds) {
                tdm_load_tile(queue16 + (size_t)(kbase + (r + 1) * KT2) * DIMD,
                              ((r + 1) & 1) ? lds1 : lds0);
                __builtin_amdgcn_s_wait_tensorcnt(1);  // round r's DMA done
            } else {
                __builtin_amdgcn_s_wait_tensorcnt(0);
            }
        }
        __syncthreads();

        const _Float16* bufp = &tile[r & 1][0];
#pragma unroll
        for (int s = 0; s < 2; ++s) {
            const int krow0 = kbase + r * KT2 + s * 16;
            // B layout (32x16): lane: N=l16, K-base = half*8
            const _Float16* bp = bufp + (s * 16 + l16) * LDP + half * 8;
            // load all B fragments first, then chain WMMAs
            v16h bf[4];
#pragma unroll
            for (int c = 0; c < 4; ++c) {
                v8h b0 = *(const v8h*)(bp + 32 * c);
                v8h b1 = *(const v8h*)(bp + 32 * c + 16);
                bf[c] = __builtin_shufflevector(b0, b1,
                    0,1,2,3,4,5,6,7,8,9,10,11,12,13,14,15);
            }
            v8f acc = {};
#pragma unroll
            for (int c = 0; c < 4; ++c) {
                acc = __builtin_amdgcn_wmma_f32_16x16x32_f16(
                    false, afrag[c], false, bf[c], (short)0, acc, false, false);
            }

            // streaming top-k: fast threshold filter
            int pred = 0;
#pragma unroll
            for (int v = 0; v < 8; ++v) pred |= (acc[v] > T[v]) ? 1 : 0;

            if (__any(pred)) {  // rare slow path (wave-uniform branch)
#pragma unroll
                for (int v = 0; v < 8; ++v) simtile[w][v + half * 8][l16] = acc[v];
                wave_lds_sync();
                if (lane < 16) {
                    const int rr = lane;
                    float th = thr[w][rr];
                    int   mp = mpos[w][rr];
#pragma unroll 4
                    for (int c2 = 0; c2 < 16; ++c2) {
                        const float val = simtile[w][rr][c2];
                        if (val > th) {
                            topv[w][rr][mp] = val;
                            topi[w][rr][mp] = krow0 + c2;
                            th = topv[w][rr][0]; mp = 0;
#pragma unroll
                            for (int j = 1; j < TOPK; ++j) {
                                const float tv = topv[w][rr][j];
                                if (tv < th) { th = tv; mp = j; }
                            }
                        }
                    }
                    thr[w][rr]  = th;
                    mpos[w][rr] = mp;
                }
                wave_lds_sync();
#pragma unroll
                for (int v = 0; v < 8; ++v) T[v] = thr[w][v + half * 8];
            }
        }
        __syncthreads();   // all waves done with buf[r&1] before DMA reuses it
    }

    if (lane < 16) {
        const int qrow = qbase + lane;
        const size_t base = (size_t)qrow * (NSPLIT * TOPK) + blockIdx.x * TOPK;
#pragma unroll
        for (int j = 0; j < TOPK; ++j) {
            pval[base + j] = topv[w][lane][j];
            pidx[base + j] = topi[w][lane][j];
        }
    }
}

// ---------------------------------------------------------------------------
// Kernel 3: merge NSPLIT*16 candidates per row -> sorted top-16 + gather.
__global__ __launch_bounds__(BLOCK) void mb_merge(
    const float* __restrict__ pval, const int* __restrict__ pidx,
    const float* __restrict__ queue,
    float* __restrict__ neigh, float* __restrict__ vals, int ncand) {
    __shared__ float sval[NSPLIT * TOPK];
    __shared__ int   sidx[NSPLIT * TOPK];
    __shared__ float rv[BLOCK];
    __shared__ int   rp[BLOCK];
    __shared__ float bv[TOPK];
    __shared__ int   bi[TOPK];

    const int row = blockIdx.x;
    const int tid = threadIdx.x;

    for (int j = tid; j < ncand; j += BLOCK) {
        sval[j] = pval[(size_t)row * ncand + j];
        sidx[j] = pidx[(size_t)row * ncand + j];
    }
    __syncthreads();

    for (int s = 0; s < TOPK; ++s) {
        float best = -1e30f; int bp = 0;
        for (int j = tid; j < ncand; j += BLOCK) {
            const float v = sval[j];
            if (v > best) { best = v; bp = j; }
        }
        rv[tid] = best; rp[tid] = bp;
        __syncthreads();
#pragma unroll
        for (int off = BLOCK / 2; off >= 1; off >>= 1) {
            if (tid < off) {
                const float v2 = rv[tid + off];
                if (v2 > rv[tid]) { rv[tid] = v2; rp[tid] = rp[tid + off]; }
            }
            __syncthreads();
        }
        if (tid == 0) {
            bv[s] = rv[0];
            bi[s] = sidx[rp[0]];
            sval[rp[0]] = -1e30f;
        }
        __syncthreads();
    }

    if (tid < TOPK) vals[(size_t)row * TOPK + tid] = bv[tid];
    for (int e = tid; e < TOPK * DIMD; e += BLOCK) {
        const int j = e >> 7, c = e & 127;
        neigh[(size_t)row * (TOPK * DIMD) + e] = queue[(size_t)bi[j] * DIMD + c];
    }
}

// ---------------------------------------------------------------------------
extern "C" void kernel_launch(void* const* d_in, const int* in_sizes, int n_in,
                              void* d_out, int out_size, void* d_ws, size_t ws_size,
                              hipStream_t stream) {
    const float* query = (const float*)d_in[0];
    const float* queue = (const float*)d_in[1];
    const int n = in_sizes[0] / DIMD;   // 2048
    const int K = in_sizes[1] / DIMD;   // 131072

    // ws: qf16 | queue16 | pval | pidx  (~49 MB total)
    char* ws = (char*)d_ws;
    size_t off = 0;
    _Float16* qf16 = (_Float16*)(ws + off);
    off += (size_t)n * DIMD * sizeof(_Float16);
    off = (off + 255) & ~(size_t)255;
    _Float16* queue16 = (_Float16*)(ws + off);
    off += (size_t)K * DIMD * sizeof(_Float16);
    off = (off + 255) & ~(size_t)255;
    float* pval = (float*)(ws + off);
    off += (size_t)n * NSPLIT * TOPK * sizeof(float);
    int* pidx = (int*)(ws + off);

    float* neigh = (float*)d_out;
    float* vals  = neigh + (size_t)n * TOPK * DIMD;

    mb_norm<<<n / (BLOCK / 32), BLOCK, 0, stream>>>(query, qf16, n);

    const size_t qelem = (size_t)K * DIMD;
    mb_cvt<<<(int)(qelem / (8 * BLOCK)), BLOCK, 0, stream>>>(queue, queue16, qelem);

    dim3 grid(NSPLIT, n / QROWS_WG);
    mb_main<<<grid, BLOCK, 0, stream>>>(qf16, queue16, pval, pidx, n, K);

    mb_merge<<<n, BLOCK, 0, stream>>>(pval, pidx, queue, neigh, vals,
                                      NSPLIT * TOPK);
}
